// MultiHeadAttention_41970420417817
// MI455X (gfx1250) — matrix operand order repaired and checked by reading, not compiled, not run on hardware
//
#include <hip/hip_runtime.h>
#include <hip/hip_bf16.h>

#define D_MODEL   1024
#define NUM_HEADS 16
#define D_K       64
#define BATCH     2
#define SEQ       2048
#define INV_SCALE 0.125f   // 1/sqrt(64)

// LDS score strip: 16 rows x SEQ, row stride padded so phase-3 row-major
// b128 reads hit distinct banks (2052 % 64 == 4 -> rows offset by 4 banks).
#define LDS_STRIDE 2052
#define SMEM_FLOATS (16 * LDS_STRIDE + 256 + 16)
#define SMEM_BYTES  (SMEM_FLOATS * 4)

typedef __attribute__((ext_vector_type(16))) _Float16 v16h;
typedef __attribute__((ext_vector_type(8)))  float    v8f;
typedef __attribute__((ext_vector_type(8)))  _Float16 h8;
typedef __attribute__((ext_vector_type(4)))  float    f4;

// ---------------------------------------------------------------------------
// Tile loaders implementing the CDNA5 WMMA VGPR layouts (ISA 7.12.2).
// A tile: 16x32 (MxK). lane<16: row=lane, halves 0..7 = K0..7, 8..15 = K16..23
//                      lane>=16: row=lane-16, halves 0..7 = K8..15, 8..15 = K24..31
// ---------------------------------------------------------------------------
__device__ __forceinline__ v16h load_a_f16(const _Float16* base, int ld, int lane) {
    const _Float16* p = base + (size_t)(lane & 15) * ld + ((lane >> 4) & 1) * 8;
    h8 lo = *(const h8*)(p);
    h8 hi = *(const h8*)(p + 16);
    v16h a;
#pragma unroll
    for (int i = 0; i < 8; ++i) { a[i] = lo[i]; a[i + 8] = hi[i]; }
    return a;
}

__device__ __forceinline__ v16h load_a_f32(const float* base, int ld, int lane) {
    const float* p = base + (size_t)(lane & 15) * ld + ((lane >> 4) & 1) * 8;
    f4 x0 = *(const f4*)(p);      f4 x1 = *(const f4*)(p + 4);
    f4 x2 = *(const f4*)(p + 16); f4 x3 = *(const f4*)(p + 20);
    v16h a;
#pragma unroll
    for (int i = 0; i < 4; ++i) {
        a[i]      = (_Float16)x0[i];
        a[i + 4]  = (_Float16)x1[i];
        a[i + 8]  = (_Float16)x2[i];
        a[i + 12] = (_Float16)x3[i];
    }
    return a;
}

// B tile: 32x16 (KxN), element (kk,n) at base[n*ld + kk].
// lane<16: col=lane, halves 0..7 = K0..7, 8..15 = K16..23 ; lane>=16: col=lane-16, halves 0..7 = K8..15, 8..15 = K24..31
__device__ __forceinline__ v16h load_b_f16(const _Float16* base, int ld, int lane) {
    const _Float16* p = base + (size_t)(lane & 15) * ld + ((lane >> 4) & 1) * 8;
    h8 lo = *(const h8*)(p);
    h8 hi = *(const h8*)(p + 16);
    v16h b;
#pragma unroll
    for (int i = 0; i < 8; ++i) { b[i] = lo[i]; b[i + 8] = hi[i]; }
    return b;
}

__device__ __forceinline__ v16h load_b_f32(const float* base, int ld, int lane) {
    const float* p = base + (size_t)(lane & 15) * ld + ((lane >> 4) & 1) * 8;
    f4 x0 = *(const f4*)(p);      f4 x1 = *(const f4*)(p + 4);
    f4 x2 = *(const f4*)(p + 16); f4 x3 = *(const f4*)(p + 20);
    v16h b;
#pragma unroll
    for (int i = 0; i < 4; ++i) {
        b[i]      = (_Float16)x0[i];
        b[i + 4]  = (_Float16)x1[i];
        b[i + 8]  = (_Float16)x2[i];
        b[i + 12] = (_Float16)x3[i];
    }
    return b;
}

// A tile sourced from the LDS score strip (fp32, LDS_STRIDE row pitch).
__device__ __forceinline__ v16h load_a_lds(const float* s, int k0, int lane) {
    const float* p = s + (lane & 15) * LDS_STRIDE + k0 + ((lane >> 4) & 1) * 8;
    f4 x0 = *(const f4*)(p);      f4 x1 = *(const f4*)(p + 4);
    f4 x2 = *(const f4*)(p + 16); f4 x3 = *(const f4*)(p + 20);
    v16h a;
#pragma unroll
    for (int i = 0; i < 4; ++i) {
        a[i]      = (_Float16)x0[i];
        a[i + 4]  = (_Float16)x1[i];
        a[i + 8]  = (_Float16)x2[i];
        a[i + 12] = (_Float16)x3[i];
    }
    return a;
}

#define WMMA_F16(a, b, c) \
    __builtin_amdgcn_wmma_f32_16x16x32_f16(false, (a), false, (b), (short)0, (c), false, false)

// ---------------------------------------------------------------------------
// Kernel 1: projection  Y = X @ W^T + bias  (fp32 in, f16 out)
// One wave per 16x64 output tile (A reused across 4 WMMAs per K-step).
//   mode 0: out[((b*H+h)*S + s)*dk + d]   (row-major per head: q, k)
//   mode 1: out[((b*H+h)*dk + d)*S + s]   (transposed per head: v)
// ---------------------------------------------------------------------------
__global__ void proj16(const float* __restrict__ X, const float* __restrict__ W,
                       const float* __restrict__ bias, _Float16* __restrict__ out,
                       int rows, int mode)
{
    const int lane = threadIdx.x & 31;
    const int wave = blockIdx.x * (blockDim.x >> 5) + (threadIdx.x >> 5);
    const int ngroups = D_MODEL / 64;                 // 16
    const int mt = wave / ngroups, ng = wave % ngroups;
    const int row0 = mt * 16, n0 = ng * 64;
    if (row0 >= rows) return;

    v8f c[4] = {};
    for (int k0 = 0; k0 < D_MODEL; k0 += 32) {
        v16h a = load_a_f32(X + (size_t)row0 * D_MODEL + k0, D_MODEL, lane);
#pragma unroll
        for (int j = 0; j < 4; ++j) {
            v16h b = load_b_f32(W + (size_t)(n0 + 16 * j) * D_MODEL + k0, D_MODEL, lane);
            c[j] = WMMA_F16(a, b, c[j]);
        }
    }

    const int col = lane & 15;
    const int rb  = ((lane >> 4) & 1) * 8;
#pragma unroll
    for (int j = 0; j < 4; ++j) {
        const int n = n0 + 16 * j + col;
        const float bvv = bias[n];
        const int h = n / D_K, d = n % D_K;
#pragma unroll
        for (int i = 0; i < 8; ++i) {
            const int r = row0 + rb + i;
            const int bidx = r / SEQ, s = r % SEQ;
            const float y = c[j][i] + bvv;
            size_t o;
            if (mode == 0)
                o = ((size_t)((bidx * NUM_HEADS + h) * SEQ + s)) * D_K + d;
            else
                o = ((size_t)((bidx * NUM_HEADS + h) * D_K + d)) * SEQ + s;
            out[o] = (_Float16)y;
        }
    }
}

// ---------------------------------------------------------------------------
// Fused attention kernel: one workgroup (256 thr, 8 waves) per 16-row query
// strip of one (b,h).  Scores -> LDS, softmax in LDS, attn streamed to HBM
// non-temporally, context accumulated from LDS.  attn touches HBM exactly once.
// ---------------------------------------------------------------------------
__global__ void attn_fused(const _Float16* __restrict__ qh, const _Float16* __restrict__ kh,
                           const _Float16* __restrict__ vt, float* __restrict__ attn,
                           _Float16* __restrict__ ctx)
{
    extern __shared__ float smem[];
    float* s_attn = smem;                       // 16 x LDS_STRIDE
    float* s_red  = smem + 16 * LDS_STRIDE;     // 256
    float* s_stat = s_red + 256;                // 16 (row max, then 1/rowsum)

    const int tid  = threadIdx.x;
    const int lane = tid & 31;
    const int wid  = tid >> 5;                  // 8 waves
    const int mtiles = SEQ / 16;                // 128 strips per (b,h)
    const int bh   = blockIdx.x / mtiles;
    const int row0 = (blockIdx.x % mtiles) * 16;
    const int b    = bh / NUM_HEADS, h = bh % NUM_HEADS;

    const _Float16* q_bh = qh + (size_t)bh * SEQ * D_K;
    const _Float16* k_bh = kh + (size_t)bh * SEQ * D_K;
    const _Float16* v_bh = vt + (size_t)bh * D_K * SEQ;

    const int col = lane & 15;
    const int rb  = ((lane >> 4) & 1) * 8;

    // ---- Phase 1: scores strip -> LDS -------------------------------------
    // A tiles (q strip, full dk) loaded once per wave.
    const v16h a0 = load_a_f16(q_bh + (size_t)row0 * D_K + 0,  D_K, lane);
    const v16h a1 = load_a_f16(q_bh + (size_t)row0 * D_K + 32, D_K, lane);

    for (int g = 0; g < 4; ++g) {               // wave covers cols [wid*256, +256)
        const int c0 = wid * 256 + g * 64;
        v8f c[4] = {};
#pragma unroll
        for (int j = 0; j < 4; ++j) {
            v16h b0 = load_b_f16(k_bh + (size_t)(c0 + 16 * j) * D_K + 0,  D_K, lane);
            c[j] = WMMA_F16(a0, b0, c[j]);
            v16h b1 = load_b_f16(k_bh + (size_t)(c0 + 16 * j) * D_K + 32, D_K, lane);
            c[j] = WMMA_F16(a1, b1, c[j]);
        }
#pragma unroll
        for (int j = 0; j < 4; ++j)
#pragma unroll
            for (int i = 0; i < 8; ++i)
                s_attn[(rb + i) * LDS_STRIDE + c0 + 16 * j + col] = c[j][i] * INV_SCALE;
    }
    __syncthreads();

    // ---- Phase 2: row softmax in LDS --------------------------------------
    const int srow = tid >> 4;                  // 16 threads per row
    const int sidx = tid & 15;
    float m = -3.402823e38f;
    for (int e = 0; e < 128; ++e)
        m = fmaxf(m, s_attn[srow * LDS_STRIDE + sidx + 16 * e]);
    s_red[tid] = m;
    __syncthreads();
    if (tid < 16) {
        float rm = s_red[tid * 16];
#pragma unroll
        for (int u = 1; u < 16; ++u) rm = fmaxf(rm, s_red[tid * 16 + u]);
        s_stat[tid] = rm;
    }
    __syncthreads();

    const float rowmax = s_stat[srow];
    float sum = 0.0f;
    for (int e = 0; e < 128; ++e) {
        float* p = &s_attn[srow * LDS_STRIDE + sidx + 16 * e];
        const float v = __expf(*p - rowmax);
        *p = v;
        sum += v;
    }
    s_red[tid] = sum;
    __syncthreads();
    if (tid < 16) {
        float rs = s_red[tid * 16];
#pragma unroll
        for (int u = 1; u < 16; ++u) rs += s_red[tid * 16 + u];
        s_stat[tid] = 1.0f / rs;
    }
    __syncthreads();

    // Normalize in LDS and stream final attn to HBM (write-once -> NT stores).
    float* attn_strip = attn + (size_t)bh * SEQ * SEQ + (size_t)row0 * SEQ;
    for (int chunk = 0; chunk < 32; ++chunk) {
        const int idx = chunk * 1024 + tid * 4;   // 16*2048 elements, f4 each
        const int r = idx >> 11, cc = idx & 2047;
        const float inv = s_stat[r];
        f4 v = *(const f4*)&s_attn[r * LDS_STRIDE + cc];
        v *= inv;
        *(f4*)&s_attn[r * LDS_STRIDE + cc] = v;
        __builtin_nontemporal_store(v, (f4*)&attn_strip[(size_t)r * SEQ + cc]);
    }
    __syncthreads();

    // ---- Phase 3: context = attn(LDS) @ v ---------------------------------
    // Wave wid owns K-segment [wid*256, wid*256+256); partial 16x64 accum.
    v8f c[4] = {};
    for (int k0 = wid * 256; k0 < wid * 256 + 256; k0 += 32) {
        v16h a = load_a_lds(s_attn, k0, lane);
#pragma unroll
        for (int j = 0; j < 4; ++j) {
            v16h bt = load_b_f16(v_bh + (size_t)(16 * j) * SEQ + k0, SEQ, lane);
            c[j] = WMMA_F16(a, bt, c[j]);
        }
    }
    __syncthreads();                            // all LDS reads done
    // Stash partials (reuse score strip region): 8 waves x 16x64 fp32.
#pragma unroll
    for (int j = 0; j < 4; ++j)
#pragma unroll
        for (int i = 0; i < 8; ++i)
            s_attn[wid * 1024 + (rb + i) * 64 + 16 * j + col] = c[j][i];
    __syncthreads();

    // Reduce 8 partials -> ctx (f16, head-merged [B,S,D] layout).
#pragma unroll
    for (int midx = 0; midx < 4; ++midx) {
        const int idx = midx * 256 + tid;       // 0..1023 over 16x64 outputs
        float acc = s_attn[idx];
#pragma unroll
        for (int w = 1; w < 8; ++w) acc += s_attn[w * 1024 + idx];
        const int r = idx >> 6, d = idx & 63;
        ctx[((size_t)(b * SEQ + row0 + r)) * D_MODEL + h * D_K + d] = (_Float16)acc;
    }
}

// ---------------------------------------------------------------------------
// Kernel 3: output = ctx @ Wo^T + bo   (f16 A, fp32 W converted, fp32 out)
// ---------------------------------------------------------------------------
__global__ void outproj16(const _Float16* __restrict__ ctx, const float* __restrict__ Wo,
                          const float* __restrict__ bo, float* __restrict__ out, int rows)
{
    const int lane = threadIdx.x & 31;
    const int wave = blockIdx.x * (blockDim.x >> 5) + (threadIdx.x >> 5);
    const int ngroups = D_MODEL / 64;                  // 16
    const int mt = wave / ngroups, ng = wave % ngroups;
    const int row0 = mt * 16, n0 = ng * 64;
    if (row0 >= rows) return;

    v8f c[4] = {};
    for (int k0 = 0; k0 < D_MODEL; k0 += 32) {
        v16h a = load_a_f16(ctx + (size_t)row0 * D_MODEL + k0, D_MODEL, lane);
#pragma unroll
        for (int j = 0; j < 4; ++j) {
            v16h b = load_b_f32(Wo + (size_t)(n0 + 16 * j) * D_MODEL + k0, D_MODEL, lane);
            c[j] = WMMA_F16(a, b, c[j]);
        }
    }

    const int col = lane & 15;
    const int rb  = ((lane >> 4) & 1) * 8;
#pragma unroll
    for (int j = 0; j < 4; ++j) {
        const float bvv = bo[n0 + 16 * j + col];
#pragma unroll
        for (int i = 0; i < 8; ++i)
            out[(size_t)(row0 + rb + i) * D_MODEL + n0 + 16 * j + col] = c[j][i] + bvv;
    }
}

// ---------------------------------------------------------------------------
extern "C" void kernel_launch(void* const* d_in, const int* in_sizes, int n_in,
                              void* d_out, int out_size, void* d_ws, size_t ws_size,
                              hipStream_t stream)
{
    const float* Q  = (const float*)d_in[0];
    const float* K  = (const float*)d_in[1];
    const float* V  = (const float*)d_in[2];
    const float* Wq = (const float*)d_in[3];
    const float* bq = (const float*)d_in[4];
    const float* Wk = (const float*)d_in[5];
    const float* bk = (const float*)d_in[6];
    const float* Wv = (const float*)d_in[7];
    const float* bv = (const float*)d_in[8];
    const float* Wo = (const float*)d_in[9];
    const float* bo = (const float*)d_in[10];

    const int BS  = BATCH * SEQ;                 // 4096 rows
    const size_t BSD = (size_t)BS * D_MODEL;     // 4,194,304 elems

    _Float16* qh  = (_Float16*)d_ws;             // [B,H,S,dk]
    _Float16* kh  = qh + BSD;                    // [B,H,S,dk]
    _Float16* vt  = kh + BSD;                    // [B,H,dk,S]
    _Float16* ctx = vt + BSD;                    // [B,S,D]

    float* out  = (float*)d_out;                 // [B,S,D]
    float* attn = out + BSD;                     // [B,H,S,S]

    const int threads = 256;                     // 8 waves per block
    const int proj_waves  = (BS / 16) * (D_MODEL / 64);        // 4096
    const int proj_blocks = proj_waves / 8;                    // 512
    const int fused_blocks = BATCH * NUM_HEADS * (SEQ / 16);   // 4096

    proj16<<<proj_blocks, threads, 0, stream>>>(Q, Wq, bq, qh, BS, 0);
    proj16<<<proj_blocks, threads, 0, stream>>>(K, Wk, bk, kh, BS, 0);
    proj16<<<proj_blocks, threads, 0, stream>>>(V, Wv, bv, vt, BS, 1);
    attn_fused<<<fused_blocks, threads, SMEM_BYTES, stream>>>(qh, kh, vt, attn, ctx);
    outproj16<<<proj_blocks, threads, 0, stream>>>(ctx, Wo, bo, out, BS);
}
